// QNetNode_3882650436611
// MI455X (gfx1250) — compile-verified, matrix operand order checked by reading, not run on hardware
//
#include <hip/hip_runtime.h>
#include <hip/hip_bf16.h>

// ---------------------------------------------------------------------------
// QNetNode structure2vec forward, CDNA5 (gfx1250) wave32 + bf16 WMMA
// - B-matrices pre-swizzled into WMMA fragment-major layout (32B/lane ->
//   two coalesced global_load_b128 per fragment)
// - SpMM kept f32 with L2-resident h/pooled (51MB each << 192MB L2)
// - readout collapsed: q = relu(h@W1a + gb) . (Wout t) + bout.t
// - full-tile fast-path epilogues (no per-element guards except last block)
// ---------------------------------------------------------------------------

typedef __attribute__((ext_vector_type(16))) __bf16 v16bf;
typedef __attribute__((ext_vector_type(8)))  float  v8f;

#define DIM 64       // embed dim D == mlp hidden H
#define FEAT 128     // input feature dim F
#define MAX_LV 2

// ---- WMMA A fragment loader (layout per CDNA5 ISA 7.12.2) -----------------
// A 16x32 bf16: lane&15 = row M; lane>>4 = h16; element e: g=e>>3, i=e&7,
//               K = g*16 + h16*8 + i.
__device__ __forceinline__ v16bf load_a_f32(const float* rowptr, int kbase,
                                            int h16, float scale) {
    v16bf a;
#pragma unroll
    for (int g = 0; g < 2; ++g) {
        const float4* p =
            reinterpret_cast<const float4*>(rowptr + kbase + g * 16 + h16 * 8);
        float4 u = p[0];
        float4 w = p[1];
        a[g * 8 + 0] = (__bf16)(u.x * scale);
        a[g * 8 + 1] = (__bf16)(u.y * scale);
        a[g * 8 + 2] = (__bf16)(u.z * scale);
        a[g * 8 + 3] = (__bf16)(u.w * scale);
        a[g * 8 + 4] = (__bf16)(w.x * scale);
        a[g * 8 + 5] = (__bf16)(w.y * scale);
        a[g * 8 + 6] = (__bf16)(w.z * scale);
        a[g * 8 + 7] = (__bf16)(w.w * scale);
    }
    return a;
}

// ---- B fragment loader: fragment-major packed layout ----------------------
__device__ __forceinline__ v16bf load_b_pk(const __bf16* pk, int frag,
                                           int lane) {
    return *reinterpret_cast<const v16bf*>(pk + ((size_t)frag * 32 + lane) * 16);
}

// ---- pack one weight matrix (K x 64, row-major f32) into fragments --------
__device__ __forceinline__ void pack_weight(const float* W, __bf16* pk,
                                            int ksteps, int tid, int nthreads) {
    int total = ksteps * 4 * 32 * 16;            // elements
    for (int idx = tid; idx < total; idx += nthreads) {
        int e    = idx & 15;
        int lane = (idx >> 4) & 31;
        int frag = idx >> 9;                     // ks*4 + t
        int ks   = frag >> 2;
        int t    = frag & 3;
        int g    = e >> 3;
        int i    = e & 7;
        int h16  = lane >> 4;
        int n    = lane & 15;
        int K    = ks * 32 + g * 16 + h16 * 8 + i;
        int col  = t * 16 + n;
        pk[idx]  = (__bf16)W[K * DIM + col];
    }
}

__global__ void convert_weights_kernel(const float* w_n2l, const float* conv_w,
                                       const float* lin1_w, __bf16* pk_n2l,
                                       __bf16* pk_conv, __bf16* pk_lin1a) {
    int tid = blockIdx.x * blockDim.x + threadIdx.x;
    int nth = gridDim.x * blockDim.x;
    pack_weight(w_n2l, pk_n2l, 4, tid, nth);     // K = 128
    pack_weight(conv_w, pk_conv, 2, tid, nth);   // K = 64
    pack_weight(lin1_w, pk_lin1a, 2, tid, nth);  // rows 0..63 of lin1_w
}

// ---- degree: deg[src] += 1 ------------------------------------------------
__global__ void degree_kernel(const int* edge_src, float* deg, int E) {
    int e = blockIdx.x * blockDim.x + threadIdx.x;
    if (e < E) atomicAdd(&deg[edge_src[e]], 1.0f);
}

// ---- input projection: x = NF @ W + b ; h = relu(x) -----------------------
__global__ void input_proj_kernel(const float* nf, const __bf16* pk,
                                  const float* bias, float* xmsg, float* hbuf,
                                  int N) {
    int wave = threadIdx.x >> 5;
    int lane = threadIdx.x & 31;
    int h16  = lane >> 4;
    int n    = lane & 15;
    int rowbase = blockIdx.x * 128 + wave * 16;

    int arow = rowbase + n;
    if (arow >= N) arow = N - 1;                 // clamp loads, guard stores
    const float* rowptr = nf + (size_t)arow * FEAT;

    v8f acc[4] = {{}, {}, {}, {}};
#pragma unroll
    for (int ks = 0; ks < 4; ++ks) {
        v16bf a = load_a_f32(rowptr, ks * 32, h16, 1.0f);
#pragma unroll
        for (int t = 0; t < 4; ++t) {
            v16bf b = load_b_pk(pk, ks * 4 + t, lane);
            acc[t] = __builtin_amdgcn_wmma_f32_16x16x32_bf16(
                false, a, false, b, (short)0, acc[t], false, false);
        }
    }
    if (rowbase + 16 <= N) {                     // full-tile fast path
#pragma unroll
        for (int t = 0; t < 4; ++t) {
            int col = t * 16 + n;
            float bv = bias[col];
#pragma unroll
            for (int r = 0; r < 8; ++r) {
                int row = rowbase + h16 * 8 + r;
                float val = acc[t][r] + bv;
                xmsg[row * DIM + col] = val;
                hbuf[row * DIM + col] = fmaxf(val, 0.0f);
            }
        }
    } else {
#pragma unroll
        for (int t = 0; t < 4; ++t) {
            int col = t * 16 + n;
            float bv = bias[col];
#pragma unroll
            for (int r = 0; r < 8; ++r) {
                int row = rowbase + h16 * 8 + r;
                if (row < N) {
                    float val = acc[t][r] + bv;
                    xmsg[row * DIM + col] = val;
                    hbuf[row * DIM + col] = fmaxf(val, 0.0f);
                }
            }
        }
    }
}

// ---- picked-node bias -----------------------------------------------------
__global__ void picked_bias_kernel(const float* bias_picked,
                                   const int* picked_idx, float* xmsg,
                                   float* hbuf) {
    int d = threadIdx.x;
    int p = picked_idx[0];
    float val = xmsg[p * DIM + d] + bias_picked[d];
    xmsg[p * DIM + d] = val;
    hbuf[p * DIM + d] = fmaxf(val, 0.0f);
}

// ---- SpMM: pooled[src,:] += h[dst,:]  (8 lanes per edge) ------------------
__global__ void spmm_kernel(const int* edge_src, const int* edge_dst,
                            const float* hbuf, float* pooled, int E) {
    int t = blockIdx.x * blockDim.x + threadIdx.x;
    int e = t >> 3;
    if (e >= E) return;
    int c   = (t & 7) * 8;
    int src = edge_src[e];
    int dst = edge_dst[e];
    const float4* hp =
        reinterpret_cast<const float4*>(hbuf + (size_t)dst * DIM + c);
    float4 u = hp[0];
    float4 w = hp[1];
    float* pp = pooled + (size_t)src * DIM + c;
    atomicAdd(pp + 0, u.x); atomicAdd(pp + 1, u.y);
    atomicAdd(pp + 2, u.z); atomicAdd(pp + 3, u.w);
    atomicAdd(pp + 4, w.x); atomicAdd(pp + 5, w.y);
    atomicAdd(pp + 6, w.z); atomicAdd(pp + 7, w.w);
}

// ---- conv round: h = relu((pooled/deg) @ conv_w + conv_b + xmsg) ----------
__global__ void conv_kernel(const float* pooled, const float* deg,
                            const __bf16* pk, const float* bias,
                            const float* xmsg, float* hbuf, int N) {
    int wave = threadIdx.x >> 5;
    int lane = threadIdx.x & 31;
    int h16  = lane >> 4;
    int n    = lane & 15;
    int rowbase = blockIdx.x * 128 + wave * 16;

    int arow = rowbase + n;
    if (arow >= N) arow = N - 1;
    float scale = 1.0f / fmaxf(deg[arow], 1.0f);   // one row per lane (A layout)
    const float* rowptr = pooled + (size_t)arow * DIM;

    v8f acc[4] = {{}, {}, {}, {}};
#pragma unroll
    for (int ks = 0; ks < 2; ++ks) {
        v16bf a = load_a_f32(rowptr, ks * 32, h16, scale);
#pragma unroll
        for (int t = 0; t < 4; ++t) {
            v16bf b = load_b_pk(pk, ks * 4 + t, lane);
            acc[t] = __builtin_amdgcn_wmma_f32_16x16x32_bf16(
                false, a, false, b, (short)0, acc[t], false, false);
        }
    }
    if (rowbase + 16 <= N) {                     // full-tile fast path
#pragma unroll
        for (int t = 0; t < 4; ++t) {
            int col = t * 16 + n;
            float bv = bias[col];
#pragma unroll
            for (int r = 0; r < 8; ++r) {
                int row = rowbase + h16 * 8 + r;
                hbuf[row * DIM + col] =
                    fmaxf(acc[t][r] + bv + xmsg[row * DIM + col], 0.0f);
            }
        }
    } else {
#pragma unroll
        for (int t = 0; t < 4; ++t) {
            int col = t * 16 + n;
            float bv = bias[col];
#pragma unroll
            for (int r = 0; r < 8; ++r) {
                int row = rowbase + h16 * 8 + r;
                if (row < N)
                    hbuf[row * DIM + col] =
                        fmaxf(acc[t][r] + bv + xmsg[row * DIM + col], 0.0f);
            }
        }
    }
}

// ---- column-sum of h for graph embedding ----------------------------------
__global__ void colsum_kernel(const float* hbuf, float* gsum, int N) {
    __shared__ float s[256];
    int tid = threadIdx.x;
    int col = tid & 63;
    int sub = tid >> 6;                          // 0..3
    float local = 0.0f;
    for (int row = blockIdx.x * 4 + sub; row < N; row += gridDim.x * 4)
        local += hbuf[(size_t)row * DIM + col];
    s[tid] = local;
    __syncthreads();
    if (tid < 64) {
        float tot = s[tid] + s[tid + 64] + s[tid + 128] + s[tid + 192];
        atomicAdd(&gsum[col], tot);
    }
}

// ---- prep: gb = lin1_b + g@lin1_w[64:] ; v = out_w @ t ; c = out_b . t ----
__global__ void prep_final_kernel(const float* gsum, const float* hbuf,
                                  const int* target_idx, const float* lin1_w,
                                  const float* lin1_b, const float* out_w,
                                  const float* out_b, float* gb, float* vvec,
                                  float* cscal, int N) {
    __shared__ float g[DIM];
    __shared__ float tv[DIM];
    int j = threadIdx.x;
    int tgt = target_idx[0];
    g[j]  = gsum[j] / (float)N;
    tv[j] = hbuf[(size_t)tgt * DIM + j];
    __syncthreads();
    float agb = lin1_b[j];
    float av  = 0.0f;
#pragma unroll 8
    for (int d = 0; d < DIM; ++d) {
        agb += g[d] * lin1_w[(DIM + d) * DIM + j];  // second-half rows of lin1_w
        av  += out_w[j * DIM + d] * tv[d];          // (out_w @ t)[j]
    }
    gb[j]   = agb;
    vvec[j] = av;
    if (j == 0) {
        float c = 0.0f;
        for (int d = 0; d < DIM; ++d) c += out_b[d] * tv[d];
        cscal[0] = c;
    }
}

// ---- fused head: q = relu(h @ lin1a + gb) . v + c -------------------------
__global__ void final_kernel(const float* hbuf, const __bf16* pk,
                             const float* gb, const float* vvec,
                             const float* cscal, float* q, int N) {
    int wave = threadIdx.x >> 5;
    int lane = threadIdx.x & 31;
    int h16  = lane >> 4;
    int n    = lane & 15;
    int rowbase = blockIdx.x * 128 + wave * 16;

    int arow = rowbase + n;
    if (arow >= N) arow = N - 1;
    const float* rowptr = hbuf + (size_t)arow * DIM;

    v8f acc[4] = {{}, {}, {}, {}};
#pragma unroll
    for (int ks = 0; ks < 2; ++ks) {
        v16bf a = load_a_f32(rowptr, ks * 32, h16, 1.0f);
#pragma unroll
        for (int t = 0; t < 4; ++t) {
            v16bf b = load_b_pk(pk, ks * 4 + t, lane);
            acc[t] = __builtin_amdgcn_wmma_f32_16x16x32_bf16(
                false, a, false, b, (short)0, acc[t], false, false);
        }
    }
    float part[8] = {0, 0, 0, 0, 0, 0, 0, 0};
#pragma unroll
    for (int t = 0; t < 4; ++t) {
        int col  = t * 16 + n;
        float bv = gb[col];
        float vv = vvec[col];
#pragma unroll
        for (int r = 0; r < 8; ++r)
            part[r] += fmaxf(acc[t][r] + bv, 0.0f) * vv;
    }
    // reduce across the 16 lanes sharing the same h16 half
#pragma unroll
    for (int m = 1; m < 16; m <<= 1)
#pragma unroll
        for (int r = 0; r < 8; ++r)
            part[r] += __shfl_xor(part[r], m, 32);
    if (n == 0) {
        float c = cscal[0];
        if (rowbase + 16 <= N) {
#pragma unroll
            for (int r = 0; r < 8; ++r)
                q[rowbase + h16 * 8 + r] = part[r] + c;
        } else {
#pragma unroll
            for (int r = 0; r < 8; ++r) {
                int row = rowbase + h16 * 8 + r;
                if (row < N) q[row] = part[r] + c;
            }
        }
    }
}

// ---------------------------------------------------------------------------
extern "C" void kernel_launch(void* const* d_in, const int* in_sizes, int n_in,
                              void* d_out, int out_size, void* d_ws,
                              size_t ws_size, hipStream_t stream) {
    const float* node_features = (const float*)d_in[0];
    const float* w_n2l         = (const float*)d_in[1];
    const float* bias_n2l      = (const float*)d_in[2];
    const float* bias_picked   = (const float*)d_in[3];
    const float* conv_w        = (const float*)d_in[4];
    const float* conv_b        = (const float*)d_in[5];
    const float* lin1_w        = (const float*)d_in[6];
    const float* lin1_b        = (const float*)d_in[7];
    const float* out_w         = (const float*)d_in[8];
    const float* out_b         = (const float*)d_in[9];
    const int*   edge_src      = (const int*)d_in[10];
    const int*   edge_dst      = (const int*)d_in[11];
    const int*   target_idx    = (const int*)d_in[12];
    const int*   picked_idx    = (const int*)d_in[13];
    float*       q             = (float*)d_out;

    const int N = in_sizes[0] / FEAT;
    const int E = in_sizes[10];

    // -------- workspace layout (256B aligned slabs) --------
    auto align256 = [](size_t x) { return (x + 255) & ~(size_t)255; };
    char*  base = (char*)d_ws;
    size_t off  = 0;
    float* deg    = (float*)(base + off); off += align256((size_t)N * 4);
    float* xmsg   = (float*)(base + off); off += align256((size_t)N * DIM * 4);
    float* hbuf   = (float*)(base + off); off += align256((size_t)N * DIM * 4);
    float* pooled = (float*)(base + off); off += align256((size_t)N * DIM * 4);
    __bf16* pk_n2l   = (__bf16*)(base + off); off += align256(FEAT * DIM * 2);
    __bf16* pk_conv  = (__bf16*)(base + off); off += align256(DIM * DIM * 2);
    __bf16* pk_lin1a = (__bf16*)(base + off); off += align256(DIM * DIM * 2);
    float* gsum  = (float*)(base + off); off += 256;
    float* gb    = (float*)(base + off); off += 256;
    float* vvec  = (float*)(base + off); off += 256;
    float* cscal = (float*)(base + off); off += 256;

    const int rowBlocks = (N + 127) / 128;

    hipMemsetAsync(deg,  0, (size_t)N * 4,  stream);
    hipMemsetAsync(gsum, 0, DIM * 4,        stream);

    convert_weights_kernel<<<8, 256, 0, stream>>>(w_n2l, conv_w, lin1_w,
                                                  pk_n2l, pk_conv, pk_lin1a);
    degree_kernel<<<(E + 255) / 256, 256, 0, stream>>>(edge_src, deg, E);

    input_proj_kernel<<<rowBlocks, 256, 0, stream>>>(node_features, pk_n2l,
                                                     bias_n2l, xmsg, hbuf, N);
    picked_bias_kernel<<<1, DIM, 0, stream>>>(bias_picked, picked_idx, xmsg,
                                              hbuf);

    for (int lv = 0; lv < MAX_LV; ++lv) {
        hipMemsetAsync(pooled, 0, (size_t)N * DIM * 4, stream);
        spmm_kernel<<<((size_t)E * 8 + 255) / 256, 256, 0, stream>>>(
            edge_src, edge_dst, hbuf, pooled, E);
        conv_kernel<<<rowBlocks, 256, 0, stream>>>(pooled, deg, pk_conv,
                                                   conv_b, xmsg, hbuf, N);
    }

    colsum_kernel<<<1024, 256, 0, stream>>>(hbuf, gsum, N);
    prep_final_kernel<<<1, DIM, 0, stream>>>(gsum, hbuf, target_idx, lin1_w,
                                             lin1_b, out_w, out_b, gb, vvec,
                                             cscal, N);
    final_kernel<<<rowBlocks, 256, 0, stream>>>(hbuf, pk_lin1a, gb, vvec,
                                                cscal, q, N);
}
